// MultiHeadLatentAttention_43413529428369
// MI455X (gfx1250) — matrix operand order repaired and checked by reading, not compiled, not run on hardware
//
#include <hip/hip_runtime.h>

// ---------------------------------------------------------------------------
// MLA forward for gfx1250 (MI455X): bf16 WMMA + async/TDM LDS staging.
// B=4, S=2048, E=2048, H=16, D=128, L=512, WIN=1024
// ---------------------------------------------------------------------------

typedef __attribute__((ext_vector_type(16))) __bf16 v16bf;
typedef __attribute__((ext_vector_type(8)))  float  v8f;
typedef unsigned int v4u __attribute__((ext_vector_type(4)));
typedef int v4i __attribute__((ext_vector_type(4)));
typedef int v8i __attribute__((ext_vector_type(8)));

union Frag16 {
  v16bf v;
  uint4 q[2];
  unsigned short us[16];
};

__device__ __forceinline__ unsigned short f2bf(float f) {
  unsigned int u = __float_as_uint(f);
  u += 0x7fffu + ((u >> 16) & 1u);           // round-to-nearest-even
  return (unsigned short)(u >> 16);
}

__device__ __forceinline__ v8f v8f_zero() {
  v8f z;
#pragma unroll
  for (int i = 0; i < 8; ++i) z[i] = 0.0f;
  return z;
}

#define WMMA_BF16(A_, B_, C_) \
  __builtin_amdgcn_wmma_f32_16x16x32_bf16(false, (A_), false, (B_), (short)0, (C_), false, false)

// --- CDNA5 async global->LDS copy (ASYNCcnt) -------------------------------
// Generic shared pointers on amdgcn are {aperture_hi32, lds_offset32}, so the
// low 32 bits of a __shared__ address are the LDS byte offset the VDST needs.
__device__ __forceinline__ void async_load_b128(void* lds_ptr, const void* gptr) {
  unsigned off = (unsigned)(size_t)lds_ptr;
  unsigned long long ga = (unsigned long long)(size_t)gptr;
  asm volatile("global_load_async_to_lds_b128 %0, %1, off"
               :: "v"(off), "v"(ga)
               : "memory");
}
__device__ __forceinline__ void wait_asynccnt0() {
  asm volatile("s_wait_asynccnt 0x0" ::: "memory");
}

// --- CDNA5 Tensor Data Mover: 2-D tile load, 8-byte elements ---------------
// Loads tile_rows rows of 256 bytes (32 x 8B) from a tensor whose row stride
// is row_elems8*8 bytes, inserting 16B of LDS padding after every 256B row
// (=> 272B LDS row stride, matching our padded [.][136] ushort tiles).
__device__ __forceinline__ void tdm_load_tile_2d(unsigned lds_byte_off,
                                                 const void* gsrc,
                                                 unsigned row_elems8,
                                                 unsigned tile_rows) {
  unsigned long long ga = (unsigned long long)(size_t)gsrc;
  v4u g0;
  g0.x = 1u;                                            // count=1 (valid D#)
  g0.y = lds_byte_off;                                  // lds_addr
  g0.z = (unsigned)(ga & 0xffffffffu);                  // global_addr[31:0]
  g0.w = (unsigned)((ga >> 32) & 0x01ffffffu) | (2u << 30);  // [56:32] | type=2
  v8i g1;
  // data_size=3 (8B) | pad_enable | pad_interval=5 (64 DW) | pad_amount=3 (4 DW)
  g1[0] = (int)((3u << 16) | (1u << 20) | (5u << 22) | (3u << 25));
  g1[1] = (int)(row_elems8 << 16);                      // tensor_dim0 lo16
  g1[2] = (int)((row_elems8 >> 16) | (tile_rows << 16));// dim0 hi | tensor_dim1 lo
  g1[3] = (int)(32u << 16);                             // dim1 hi=0 | tile_dim0=32
  g1[4] = (int)tile_rows;                               // tile_dim1 | tile_dim2=0
  g1[5] = (int)row_elems8;                              // tensor_dim0_stride lo32
  g1[6] = 0;                                            // stride hi | dim1_stride lo
  g1[7] = 0;
  v4i z4 = {0, 0, 0, 0};
#if defined(__clang_major__) && (__clang_major__ >= 23)
  v8i z8 = {0, 0, 0, 0, 0, 0, 0, 0};
  __builtin_amdgcn_tensor_load_to_lds(g0, g1, z4, z4, z8, 0);
#else
  __builtin_amdgcn_tensor_load_to_lds(g0, g1, z4, z4, 0);
#endif
}

// ---------------------------------------------------------------------------
// fp32 -> bf16 conversion, 4 elements/thread
// ---------------------------------------------------------------------------
__global__ __launch_bounds__(256) void cvt_f32_bf16_kernel(
    const float* __restrict__ in, unsigned short* __restrict__ out, long long n4) {
  long long i = (long long)blockIdx.x * blockDim.x + threadIdx.x;
  if (i < n4) {
    float4 f = ((const float4*)in)[i];
    uint2 p;
    p.x = (unsigned)f2bf(f.x) | ((unsigned)f2bf(f.y) << 16);
    p.y = (unsigned)f2bf(f.z) | ((unsigned)f2bf(f.w) << 16);
    ((uint2*)out)[i] = p;
  }
}

// ---------------------------------------------------------------------------
// C[M,N] = A[M,K](bf16) @ W[N,K](bf16)^T + bias  -- 128x128x32 tiles, 8 waves.
// Requires M%128==0, N%128==0, K%32==0.
// ---------------------------------------------------------------------------
template <bool BF16_OUT>
__global__ __launch_bounds__(256) void gemm_bf16_kernel(
    const unsigned short* __restrict__ A, const unsigned short* __restrict__ W,
    const float* __restrict__ bias, void* __restrict__ Cout,
    int M, int N, int K) {
  __shared__ unsigned short Alds[128][40];    // [m][k], +8 pad (80B rows, 16B aligned)
  __shared__ unsigned short Blds[32][136];    // [k][n] transposed, +8 pad (272B rows)

  const int tid  = threadIdx.x;
  const int lane = tid & 31;
  const int wave = tid >> 5;
  const int wm = wave >> 2;                   // 0..1 -> 64-row slab
  const int wn = wave & 3;                    // 0..3 -> 32-col slab
  const int m0 = blockIdx.y * 128;
  const int n0 = blockIdx.x * 128;

  v8f acc[4][2];
#pragma unroll
  for (int i = 0; i < 4; ++i)
#pragma unroll
    for (int j = 0; j < 2; ++j) acc[i][j] = v8f_zero();

  const int a_row = tid >> 1;                 // 0..127
  const int a_col = (tid & 1) * 16;           // 0 or 16
  const unsigned short* ag = A + (size_t)(m0 + a_row) * K + a_col;
  const unsigned short* wg = W + (size_t)(n0 + a_row) * K + a_col;

  const int halfSel = (lane >= 16) ? 8 : 0;
  const int arow_f  = wm * 64 + (lane & 15);

  for (int k0 = 0; k0 < K; k0 += 32) {
    __syncthreads();
    // A tile: async global->LDS copy (no VGPR round-trip)
    async_load_b128(&Alds[a_row][a_col],     ag + k0);
    async_load_b128(&Alds[a_row][a_col + 8], ag + k0 + 8);
    // W tile: through VGPRs (needs the [n][k] -> [k][n] transpose)
    uint4 wv0 = *(const uint4*)(wg + k0);
    uint4 wv1 = *(const uint4*)(wg + k0 + 8);
    if (k0 + 32 < K) {
      __builtin_prefetch(ag + k0 + 32, 0, 1);
      __builtin_prefetch(wg + k0 + 32, 0, 1);
    }
    {
      Frag16 t; t.q[0] = wv0; t.q[1] = wv1;
#pragma unroll
      for (int i = 0; i < 16; ++i) Blds[a_col + i][a_row] = t.us[i];
    }
    wait_asynccnt0();
    __syncthreads();

    // A fragment: lanes 0-15 = rows, K slots {hs..hs+7, 16+hs..16+hs+7}
    Frag16 af[4];
#pragma unroll
    for (int i = 0; i < 4; ++i) {
      const unsigned short* p = &Alds[arow_f + i * 16][halfSel];
      af[i].q[0] = *(const uint4*)p;
      af[i].q[1] = *(const uint4*)(p + 16);
    }
    // B fragment: N = lane & 15, K slots {hs..hs+7, 16+hs..16+hs+7}
    Frag16 bf[2];
#pragma unroll
    for (int j = 0; j < 2; ++j) {
#pragma unroll
      for (int s = 0; s < 16; ++s)
        bf[j].us[s] = Blds[(s < 8 ? s : s + 8) + halfSel][wn * 32 + j * 16 + (lane & 15)];
    }
#pragma unroll
    for (int i = 0; i < 4; ++i)
#pragma unroll
      for (int j = 0; j < 2; ++j)
        acc[i][j] = WMMA_BF16(af[i].v, bf[j].v, acc[i][j]);
  }

  const int rowHi = (lane >= 16) ? 8 : 0;
#pragma unroll
  for (int j = 0; j < 2; ++j) {
    const int n = n0 + wn * 32 + j * 16 + (lane & 15);
    const float bsv = bias[n];
#pragma unroll
    for (int i = 0; i < 4; ++i) {
#pragma unroll
      for (int r = 0; r < 8; ++r) {
        const int m = m0 + wm * 64 + i * 16 + r + rowHi;
        const float val = acc[i][j][r] + bsv;
        if (BF16_OUT) ((unsigned short*)Cout)[(size_t)m * N + n] = f2bf(val);
        else          ((float*)Cout)[(size_t)m * N + n] = val;
      }
    }
  }
}

// ---------------------------------------------------------------------------
// LayerNorm over L=512, one block (256 threads) per row; bf16 output.
// ---------------------------------------------------------------------------
__global__ __launch_bounds__(256) void layernorm_kernel(
    const float* __restrict__ in, const float* __restrict__ g,
    const float* __restrict__ bb, unsigned short* __restrict__ out) {
  const int L = 512;
  const int row = blockIdx.x;
  const int tid = threadIdx.x;
  const float* x = in + (size_t)row * L;
  float v0 = x[tid], v1 = x[tid + 256];
  float s  = v0 + v1;
  float ss = v0 * v0 + v1 * v1;
#pragma unroll
  for (int off = 16; off > 0; off >>= 1) {
    s  += __shfl_xor(s,  off, 32);
    ss += __shfl_xor(ss, off, 32);
  }
  __shared__ float sh_s[8], sh_q[8], sh_res[2];
  const int lane = tid & 31, wave = tid >> 5;
  if (lane == 0) { sh_s[wave] = s; sh_q[wave] = ss; }
  __syncthreads();
  if (tid == 0) {
    float S = 0.f, Q = 0.f;
    for (int i = 0; i < 8; ++i) { S += sh_s[i]; Q += sh_q[i]; }
    float mu  = S / L;
    float var = Q / L - mu * mu;
    sh_res[0] = mu;
    sh_res[1] = rsqrtf(var + 1e-5f);
  }
  __syncthreads();
  const float mu = sh_res[0], rs = sh_res[1];
  out[(size_t)row * L + tid]       = f2bf((v0 - mu) * rs * g[tid]       + bb[tid]);
  out[(size_t)row * L + tid + 256] = f2bf((v1 - mu) * rs * g[tid + 256] + bb[tid + 256]);
}

// ---------------------------------------------------------------------------
// Flash attention with causal + sliding-window mask.
// grid = (S/128, H, B); block = 256 (8 waves); wave w -> 16-query tile.
// K/V tiles (32 x 256B rows, stride 4096B) are DMA'd by the Tensor Data Mover
// into padded LDS (272B row stride via TDM pad fields). S^T = K @ Q^T so
// softmax stats are per-lane and S^T's C-layout IS P's bf16 A-fragment layout.
// ---------------------------------------------------------------------------
__global__ __launch_bounds__(256) void mla_attention_kernel(
    const unsigned short* __restrict__ Qb, const unsigned short* __restrict__ Kb,
    const unsigned short* __restrict__ Vb, unsigned short* __restrict__ Ob) {
  const int S = 2048, E = 2048, WIN = 1024;
  __shared__ unsigned short Klds[32][136];    // [key][d], 272B row stride
  __shared__ unsigned short Vlds[32][136];

  const int tid = threadIdx.x, lane = tid & 31, wave = tid >> 5;
  const int qb = blockIdx.x, h = blockIdx.y, batch = blockIdx.z;
  const int q0 = qb * 128 + wave * 16;
  const size_t base = (size_t)batch * S * E + (size_t)h * 128;

  // Q^T fragments (B operand): lane & 15 = query, slots = d in K-slot order {hs..hs+7, 16+hs..16+hs+7}. Loaded once.
  Frag16 qf[4];
#pragma unroll
  for (int ks = 0; ks < 4; ++ks)
#pragma unroll
    for (int j = 0; j < 16; ++j)
      qf[ks].us[j] = Qb[base + (size_t)(q0 + (lane & 15)) * E + ks * 32 + (j < 8 ? j : j + 8) + ((lane >> 4) << 3)];

  v8f oacc[8];
#pragma unroll
  for (int dt = 0; dt < 8; ++dt) oacc[dt] = v8f_zero();
  float m_run = -1e30f, l_run = 0.0f;

  const int halfSel = (lane >= 16) ? 8 : 0;
  const int srcBase = (lane >> 4) << 3;           // 8*(lane/16)
  const int qrow = q0 + (lane & 15);
  const float sc = 0.08838834764831845f;          // 1/sqrt(128)
  const unsigned klds_off = (unsigned)(size_t)&Klds[0][0];
  const unsigned vlds_off = (unsigned)(size_t)&Vlds[0][0];

  int lo = qb * 128 - (WIN - 1);
  if (lo < 0) lo = 0;
  lo &= ~31;
  const int hi = qb * 128 + 128;

  for (int kt = lo; kt < hi; kt += 32) {
    __syncthreads();
    if (wave == 0) {
      tdm_load_tile_2d(klds_off, Kb + base + (size_t)kt * E, E / 4, 32);
      __builtin_amdgcn_s_wait_tensorcnt(0);
    } else if (wave == 1) {
      tdm_load_tile_2d(vlds_off, Vb + base + (size_t)kt * E, E / 4, 32);
      __builtin_amdgcn_s_wait_tensorcnt(0);
    }
    __syncthreads();

    // S^T[key][q] over 32 keys x 16 queries, K-dim = D = 128 (4 steps)
    v8f st0 = v8f_zero(), st1 = v8f_zero();
#pragma unroll
    for (int ks = 0; ks < 4; ++ks) {
      Frag16 a0f, a1f;
      const unsigned short* p0 = &Klds[lane & 15][ks * 32 + halfSel];
      const unsigned short* p1 = &Klds[16 + (lane & 15)][ks * 32 + halfSel];
      a0f.q[0] = *(const uint4*)p0; a0f.q[1] = *(const uint4*)(p0 + 16);
      a1f.q[0] = *(const uint4*)p1; a1f.q[1] = *(const uint4*)(p1 + 16);
      st0 = WMMA_BF16(a0f.v, qf[ks].v, st0);
      st1 = WMMA_BF16(a1f.v, qf[ks].v, st1);
    }

    // scale + causal/window mask; per-lane (per-query) running max
    float sv[16];
    float mloc = -__builtin_inff();
#pragma unroll
    for (int mt = 0; mt < 2; ++mt)
#pragma unroll
      for (int r = 0; r < 8; ++r) {
        const int key = kt + mt * 16 + r + srcBase;
        float s = (mt ? st1[r] : st0[r]) * sc;
        const bool ok = (key <= qrow) && (qrow - key < WIN);
        s = ok ? s : -__builtin_inff();
        sv[mt * 8 + r] = s;
        mloc = fmaxf(mloc, s);
      }
    // merge lane halves (lanes l <-> l^16 hold the other 16 keys of this query)
    float mo = __int_as_float(__builtin_amdgcn_ds_swizzle(__float_as_int(mloc), 0x401f));
    const float mtile = fmaxf(fmaxf(mloc, mo), -1e30f);
    const float mnew  = fmaxf(m_run, mtile);
    const float alpha = __expf(m_run - mnew);

    // P = exp(S - m): S^T C-layout == bf16 A-fragment layout -> pack in-register
    Frag16 pf;
    float lloc = 0.0f;
#pragma unroll
    for (int i = 0; i < 16; ++i) {
      const float e = __expf(sv[i] - mnew);
      lloc += e;
      pf.us[i] = f2bf(e);
    }
    const float lot = __int_as_float(__builtin_amdgcn_ds_swizzle(__float_as_int(lloc), 0x401f));
    l_run = l_run * alpha + lloc + lot;
    m_run = mnew;

    // broadcast alpha per output slot row (q = r + srcBase) via bpermute
    float alphas[8];
#pragma unroll
    for (int r = 0; r < 8; ++r)
      alphas[r] = __int_as_float(
          __builtin_amdgcn_ds_bpermute((r + srcBase) * 4, __float_as_int(alpha)));

    // O = O*alpha + P @ V   (V B-fragment: lane & 15 = d, slots = keys in K-slot order)
#pragma unroll
    for (int dt = 0; dt < 8; ++dt) {
#pragma unroll
      for (int r = 0; r < 8; ++r) oacc[dt][r] *= alphas[r];
      Frag16 vf;
#pragma unroll
      for (int s = 0; s < 16; ++s)
        vf.us[s] = Vlds[(s < 8 ? s : s + 8) + halfSel][dt * 16 + (lane & 15)];
      oacc[dt] = WMMA_BF16(pf.v, vf.v, oacc[dt]);
    }
  }

  const float rcp = 1.0f / l_run;
  float scales[8];
#pragma unroll
  for (int r = 0; r < 8; ++r)
    scales[r] = __int_as_float(
        __builtin_amdgcn_ds_bpermute((r + srcBase) * 4, __float_as_int(rcp)));
#pragma unroll
  for (int dt = 0; dt < 8; ++dt)
#pragma unroll
    for (int r = 0; r < 8; ++r) {
      const int q = q0 + r + srcBase;
      Ob[base + (size_t)q * E + dt * 16 + (lane & 15)] = f2bf(oacc[dt][r] * scales[r]);
    }
}

// ---------------------------------------------------------------------------
// Orchestration
// ---------------------------------------------------------------------------
extern "C" void kernel_launch(void* const* d_in, const int* in_sizes, int n_in,
                              void* d_out, int out_size, void* d_ws, size_t ws_size,
                              hipStream_t stream) {
  (void)in_sizes; (void)n_in; (void)out_size; (void)ws_size;
  const int Bb = 4, S = 2048, E = 2048, H = 16, Ll = 512;
  const size_t BS  = (size_t)Bb * S;     // 8192 rows
  const size_t BSE = BS * E;             // 16,777,216

  const float* x     = (const float*)d_in[0];
  const float* w_lat = (const float*)d_in[1];
  const float* b_lat = (const float*)d_in[2];
  const float* g_ln  = (const float*)d_in[3];
  const float* b_ln  = (const float*)d_in[4];
  const float* w_kv  = (const float*)d_in[5];
  const float* b_kv  = (const float*)d_in[6];
  const float* w_q   = (const float*)d_in[7];
  const float* b_q   = (const float*)d_in[8];
  const float* w_o   = (const float*)d_in[9];
  const float* b_o   = (const float*)d_in[10];
  float* out = (float*)d_out;

  char* ws = (char*)d_ws;
  auto alloc = [&](size_t bytes) -> char* {
    char* p = ws;
    ws += (bytes + 255) & ~(size_t)255;
    return p;
  };
  unsigned short* x_bf    = (unsigned short*)alloc(BSE * 2);
  unsigned short* wq_bf   = (unsigned short*)alloc((size_t)E * E * 2);
  unsigned short* wlat_bf = (unsigned short*)alloc((size_t)Ll * E * 2);
  unsigned short* wkv_bf  = (unsigned short*)alloc((size_t)2 * E * Ll * 2);
  unsigned short* wo_bf   = (unsigned short*)alloc((size_t)E * E * 2);
  unsigned short* q_bf    = (unsigned short*)alloc(BSE * 2);
  unsigned short* k_bf    = (unsigned short*)alloc(BSE * 2);
  unsigned short* v_bf    = (unsigned short*)alloc(BSE * 2);
  unsigned short* attn_bf = (unsigned short*)alloc(BSE * 2);
  float*          lat_f   = (float*)alloc(BS * Ll * 4);
  unsigned short* lat_bf  = (unsigned short*)alloc(BS * Ll * 2);

  auto cvt = [&](const float* src, unsigned short* dst, size_t n) {
    long long n4 = (long long)(n / 4);
    cvt_f32_bf16_kernel<<<(unsigned)((n4 + 255) / 256), 256, 0, stream>>>(src, dst, n4);
  };
  cvt(x,     x_bf,    BSE);
  cvt(w_q,   wq_bf,   (size_t)E * E);
  cvt(w_lat, wlat_bf, (size_t)Ll * E);
  cvt(w_kv,  wkv_bf,  (size_t)2 * E * Ll);
  cvt(w_o,   wo_bf,   (size_t)E * E);

  // q = x @ w_q^T + b_q                       (8192 x 2048, K=2048) -> bf16
  gemm_bf16_kernel<true><<<dim3(E / 128, (unsigned)(BS / 128)), 256, 0, stream>>>(
      x_bf, wq_bf, b_q, q_bf, (int)BS, E, E);

  // latent_pre = x @ w_lat^T + b_lat          (8192 x 512, K=2048) -> f32
  gemm_bf16_kernel<false><<<dim3(Ll / 128, (unsigned)(BS / 128)), 256, 0, stream>>>(
      x_bf, wlat_bf, b_lat, lat_f, (int)BS, Ll, E);

  // latent = LN(latent_pre) -> bf16
  layernorm_kernel<<<(unsigned)BS, 256, 0, stream>>>(lat_f, g_ln, b_ln, lat_bf);

  // k = latent @ w_kv[:E]^T + b_kv[:E];  v = latent @ w_kv[E:]^T + b_kv[E:]
  gemm_bf16_kernel<true><<<dim3(E / 128, (unsigned)(BS / 128)), 256, 0, stream>>>(
      lat_bf, wkv_bf, b_kv, k_bf, (int)BS, E, Ll);
  gemm_bf16_kernel<true><<<dim3(E / 128, (unsigned)(BS / 128)), 256, 0, stream>>>(
      lat_bf, wkv_bf + (size_t)E * Ll, b_kv + E, v_bf, (int)BS, E, Ll);

  // attention (causal + sliding window) -> bf16 (b, s, E)
  mla_attention_kernel<<<dim3(S / 128, H, Bb), 256, 0, stream>>>(q_bf, k_bf, v_bf, attn_bf);

  // out = attn @ w_o^T + b_o -> f32
  gemm_bf16_kernel<false><<<dim3(E / 128, (unsigned)(BS / 128)), 256, 0, stream>>>(
      attn_bf, wo_bf, b_o, out, (int)BS, E, E);
}
